// LSTM_35751307772702
// MI455X (gfx1250) — compile-verified, matrix operand order checked by reading, not run on hardware
//
#include <hip/hip_runtime.h>

typedef __bf16 bf16;
typedef __attribute__((ext_vector_type(16))) __bf16 v16bf;
typedef __attribute__((ext_vector_type(8)))  float  v8f;

#define HDIM  1152
#define G4H   4608
#define SEQ   256
#define BATCH 16
#define ROWS  4096      // BATCH*SEQ
#define E_IN  400
#define E_PAD 416
#define VOCAB 33278
#define REC_WGS 72      // 1152/16 column groups

union Frag16 { v16bf v; uint4 q[2]; };

__device__ __forceinline__ v8f wmma_bf16(v16bf a, v16bf b, v8f c) {
  // D = A(16x32) * B(32x16) + C, fp32 accum
  return __builtin_amdgcn_wmma_f32_16x16x32_bf16(false, a, false, b, (short)0, c,
                                                 false, false);
}

__device__ __forceinline__ v16bf load_frag_g(const bf16* p) {
  // 16-bit A/B fragment: k = {0..7} and {16..23} (+8 for upper lane half),
  // both contiguous in a K-major row -> two b128 loads.
  Frag16 f;
  f.q[0] = *(const uint4*)p;
  f.q[1] = *(const uint4*)(p + 16);
  return f.v;
}

__device__ __forceinline__ float sigmoid_f(float x) {
  return 1.0f / (1.0f + __expf(-x));
}

// CDNA5 async copy: LDS[vdst_lane] = MEM[vaddr_lane], tracked by ASYNCcnt.
__device__ __forceinline__ void async_g2l_b128(const bf16* gptr, const bf16* lptr) {
  unsigned           laddr = (unsigned)(size_t)lptr;          // LDS byte address
  unsigned long long gaddr = (unsigned long long)(size_t)gptr;
  asm volatile("global_load_async_to_lds_b128 %0, %1, off"
               :: "v"(laddr), "v"(gaddr) : "memory");
}

// ---------------------------------------------------------------------------
// Generic bf16 GEMM: C[M,N] = A[M,K] * BT[N,K]^T + bias, fp32 out.
// 256 threads = 8 waves; block tile 128x256; wave tile 64x64 (4x4 WMMA tiles).
// Double-buffered GLOBAL_LOAD_ASYNC_TO_LDS_B128 staging pipelined on ASYNCcnt:
// async loads complete in order, so s_wait_asynccnt 6 (= next chunk's issues)
// proves the current chunk has landed while the next streams behind the WMMAs.
// ---------------------------------------------------------------------------
__global__ __launch_bounds__(256) void gemm_bf16_tn_kernel(
    const bf16* __restrict__ A, int lda,
    const bf16* __restrict__ BT, int ldb,
    const float* __restrict__ bias,
    float* __restrict__ C, int ldc,
    int N, int K) {
  __shared__ bf16 lA[2][128 * 40];   // stride 40 -> conflict-free b128 DS reads
  __shared__ bf16 lB[2][256 * 40];
  __shared__ bf16 sink[8];           // async sink for N-tail lanes (keeps cnt uniform)

  const int tid  = threadIdx.x;
  const int lane = tid & 31;
  const int wid  = tid >> 5;
  const int wm   = wid >> 2;          // 0..1  (64-row wave tile)
  const int wn   = wid & 3;           // 0..3  (64-col wave tile)
  const int ln   = lane & 15;
  const int hlf  = lane >> 4;
  const int bm0  = blockIdx.y * 128;
  const int bn0  = blockIdx.x * 256;

  // Zero-fill B rows beyond N once: those LDS rows are never written by the
  // async stream (their lanes target the sink), so they stay zero (= padding).
#pragma unroll
  for (int b = 0; b < 2; b++)
#pragma unroll
    for (int i = 0; i < 4; i++) {
      int j   = i * 256 + tid;
      int row = j >> 2;
      int cq  = (j & 3) * 8;
      if (bn0 + row >= N) {
        uint4 z; z.x = z.y = z.z = z.w = 0u;
        *(uint4*)&lB[b][row * 40 + cq] = z;
      }
    }

  v8f acc[4][4];
#pragma unroll
  for (int mt = 0; mt < 4; mt++)
#pragma unroll
    for (int nt = 0; nt < 4; nt++)
#pragma unroll
      for (int e = 0; e < 8; e++) acc[mt][nt][e] = 0.0f;

  // stage one K-chunk (A 128x32 + BT 256x32): 1536 b128 units / 256 thr = 6 each
  auto stage = [&](int kc, int p) {
#pragma unroll
    for (int i = 0; i < 6; i++) {
      int idx = i * 256 + tid;
      if (idx < 512) {                 // A tile
        int row = idx >> 2;
        int cq  = (idx & 3) * 8;
        async_g2l_b128(&A[(size_t)(bm0 + row) * lda + kc + cq],
                       &lA[p][row * 40 + cq]);
      } else {                         // B tile (N-tail lanes -> sink)
        int j   = idx - 512;
        int row = j >> 2;
        int cq  = (j & 3) * 8;
        int gr  = bn0 + row;
        const bf16* g = (gr < N) ? &BT[(size_t)gr * ldb + kc + cq] : &BT[kc + cq];
        bf16*       l = (gr < N) ? &lB[p][row * 40 + cq] : &sink[0];
        async_g2l_b128(g, l);
      }
    }
  };

  const int KCH = K >> 5;
  stage(0, 0);
  for (int ic = 0; ic < KCH; ic++) {
    const int p = ic & 1;
    if (ic + 1 < KCH) {
      stage((ic + 1) * 32, 1 - p);                       // prefetch next chunk
      asm volatile("s_wait_asynccnt 0x6" ::: "memory");  // chunk ic has landed
    } else {
      asm volatile("s_wait_asynccnt 0x0" ::: "memory");
    }
    __syncthreads();

    v16bf fb[4];
#pragma unroll
    for (int nt = 0; nt < 4; nt++) {
      int r = wn * 64 + nt * 16 + ln;
      Frag16 f;
      f.q[0] = *(const uint4*)&lB[p][r * 40 + 8 * hlf];
      f.q[1] = *(const uint4*)&lB[p][r * 40 + 16 + 8 * hlf];
      fb[nt] = f.v;
    }
#pragma unroll
    for (int mt = 0; mt < 4; mt++) {
      int r = wm * 64 + mt * 16 + ln;
      Frag16 fa;
      fa.q[0] = *(const uint4*)&lA[p][r * 40 + 8 * hlf];
      fa.q[1] = *(const uint4*)&lA[p][r * 40 + 16 + 8 * hlf];
#pragma unroll
      for (int nt = 0; nt < 4; nt++)
        acc[mt][nt] = wmma_bf16(fa.v, fb[nt], acc[mt][nt]);
    }
    __syncthreads();   // DS reads of buf p done before it is re-staged
  }

  // epilogue: D layout -> lane n = lane&15, row m = j + 8*(lane>>4)
#pragma unroll
  for (int nt = 0; nt < 4; nt++) {
    int col = bn0 + wn * 64 + nt * 16 + ln;
    if (col < N) {
      float bv = bias[col];
#pragma unroll
      for (int mt = 0; mt < 4; mt++) {
        int r0 = bm0 + wm * 64 + mt * 16 + 8 * hlf;
#pragma unroll
        for (int j = 0; j < 8; j++)
          C[(size_t)(r0 + j) * ldc + col] = acc[mt][nt][j] + bv;
      }
    }
  }
}

// ---------------------------------------------------------------------------
// Persistent LSTM layer: 72 single-wave WGs, each owns 16 h-columns.
// Per step: gates[:,cols] = xp[:,t,cols] + h_{t-1} @ w_hh[:,cols] via WMMA,
// fused sigmoid/tanh/cell update, grid barrier between steps.
// ---------------------------------------------------------------------------
__global__ __launch_bounds__(32) void lstm_layer_kernel(
    const float* __restrict__ xp,      // [ROWS, 4H] row = b*SEQ + t
    const bf16*  __restrict__ whhT,    // [4H, H]  (transposed w_hh, K-contig)
    bf16* __restrict__ hbufA,          // [BATCH, H] ping
    bf16* __restrict__ hbufB,          // [BATCH, H] pong
    bf16* __restrict__ y,              // [ROWS, H]
    float* __restrict__ h_final,       // [BATCH, H]
    float* __restrict__ c_final,       // [BATCH, H]
    int* __restrict__ cnt) {           // [SEQ] arrive counters (pre-zeroed)
  const int lane = threadIdx.x & 31;
  const int ln   = lane & 15;
  const int hlf  = lane >> 4;
  const int col0 = blockIdx.x * 16;

  v8f c_st;
#pragma unroll
  for (int e = 0; e < 8; e++) c_st[e] = 0.0f;

  for (int t = 0; t < SEQ; t++) {
    const bf16* hr = (t & 1) ? hbufB : hbufA;
    bf16*       hw = (t & 1) ? hbufA : hbufB;

    v8f ai, af, ag, ao;
#pragma unroll
    for (int e = 0; e < 8; e++) { ai[e] = 0.f; af[e] = 0.f; ag[e] = 0.f; ao[e] = 0.f; }

    for (int kc = 0; kc < HDIM; kc += 32) {
      v16bf a   = load_frag_g(&hr[ln * HDIM + kc + 8 * hlf]);
      v16bf bi  = load_frag_g(&whhT[(size_t)(0 * HDIM + col0 + ln) * HDIM + kc + 8 * hlf]);
      v16bf bf2 = load_frag_g(&whhT[(size_t)(1 * HDIM + col0 + ln) * HDIM + kc + 8 * hlf]);
      v16bf bg  = load_frag_g(&whhT[(size_t)(2 * HDIM + col0 + ln) * HDIM + kc + 8 * hlf]);
      v16bf bo  = load_frag_g(&whhT[(size_t)(3 * HDIM + col0 + ln) * HDIM + kc + 8 * hlf]);
      ai = wmma_bf16(a, bi,  ai);
      af = wmma_bf16(a, bf2, af);
      ag = wmma_bf16(a, bg,  ag);
      ao = wmma_bf16(a, bo,  ao);
    }

#pragma unroll
    for (int j = 0; j < 8; j++) {
      int m = j + 8 * hlf;                         // batch row of this element
      size_t xr = ((size_t)(m * SEQ + t)) * G4H + col0 + ln;
      float gi = sigmoid_f(ai[j] + xp[xr + 0 * HDIM]);
      float gf = sigmoid_f(af[j] + xp[xr + 1 * HDIM]);
      float gg = tanhf    (ag[j] + xp[xr + 2 * HDIM]);
      float go = sigmoid_f(ao[j] + xp[xr + 3 * HDIM]);
      float cv = gf * c_st[j] + gi * gg;
      c_st[j] = cv;
      float hv = go * tanhf(cv);
      hw[m * HDIM + col0 + ln] = (bf16)hv;
      y[((size_t)(m * SEQ + t)) * HDIM + col0 + ln] = (bf16)hv;
      if (t == SEQ - 1) {
        h_final[m * HDIM + col0 + ln] = hv;
        c_final[m * HDIM + col0 + ln] = cv;
      }
    }

    if (t < SEQ - 1) {
      if (lane == 0) {
        __hip_atomic_fetch_add(&cnt[t], 1, __ATOMIC_RELEASE, __HIP_MEMORY_SCOPE_AGENT);
        while (__hip_atomic_load(&cnt[t], __ATOMIC_ACQUIRE, __HIP_MEMORY_SCOPE_AGENT) < REC_WGS)
          __builtin_amdgcn_s_sleep(2);
      }
      __builtin_amdgcn_wave_barrier();
    }
  }
}

// ---------------------------------------------------------------------------
// Conversion / layout kernels
// ---------------------------------------------------------------------------
__global__ void embed_kernel(const int* __restrict__ x, const float* __restrict__ emb,
                             bf16* __restrict__ e) {
  int idx = blockIdx.x * blockDim.x + threadIdx.x;
  if (idx >= ROWS * E_PAD) return;
  int r = idx / E_PAD, c = idx % E_PAD;
  float v = 0.f;
  if (c < E_IN) {
    int tok = x[r];
    v = emb[(size_t)tok * E_IN + c];
  }
  e[idx] = (bf16)v;
}

// out[n*Kout + k] = (k < Kin) ? in[k*N + n] : 0      (fp32 [Kin,N] -> bf16 [N,Kout])
__global__ void transpose_cvt_kernel(const float* __restrict__ in, bf16* __restrict__ out,
                                     int Kin, int Kout, int N) {
  long idx = (long)blockIdx.x * blockDim.x + threadIdx.x;
  if (idx >= (long)N * Kout) return;
  int n = (int)(idx / Kout), k = (int)(idx % Kout);
  float v = (k < Kin) ? in[(size_t)k * N + n] : 0.f;
  out[idx] = (bf16)v;
}

__global__ void cvt_kernel(const float* __restrict__ in, bf16* __restrict__ out, long total) {
  long idx = (long)blockIdx.x * blockDim.x + threadIdx.x;
  if (idx < total) out[idx] = (bf16)in[idx];
}

// ---------------------------------------------------------------------------
extern "C" void kernel_launch(void* const* d_in, const int* in_sizes, int n_in,
                              void* d_out, int out_size, void* d_ws, size_t ws_size,
                              hipStream_t stream) {
  (void)in_sizes; (void)n_in; (void)out_size; (void)ws_size;

  const int*   x     = (const int*)  d_in[0];
  const float* emb   = (const float*)d_in[1];
  const float* w_ih0 = (const float*)d_in[2];
  const float* w_hh0 = (const float*)d_in[3];
  const float* b0    = (const float*)d_in[4];
  const float* w_ih1 = (const float*)d_in[5];
  const float* w_hh1 = (const float*)d_in[6];
  const float* b1    = (const float*)d_in[7];
  const float* fc_w  = (const float*)d_in[8];
  const float* fc_b  = (const float*)d_in[9];
  float* out = (float*)d_out;

  char* base = (char*)d_ws;
  size_t off = 0;
  auto carve = [&](size_t bytes) -> char* {
    char* p = base + off;
    off += (bytes + 255) & ~(size_t)255;
    return p;
  };
  bf16* eB    = (bf16*)carve((size_t)ROWS * E_PAD * 2);
  bf16* wih0T = (bf16*)carve((size_t)G4H * E_PAD * 2);
  bf16* whh0T = (bf16*)carve((size_t)G4H * HDIM * 2);
  bf16* wih1T = (bf16*)carve((size_t)G4H * HDIM * 2);
  bf16* whh1T = (bf16*)carve((size_t)G4H * HDIM * 2);
  bf16* fcwB  = (bf16*)carve((size_t)VOCAB * HDIM * 2);
  float* xp   = (float*)carve((size_t)ROWS * G4H * 4);     // reused for both layers
  bf16* y0    = (bf16*)carve((size_t)ROWS * HDIM * 2);
  bf16* y1    = (bf16*)carve((size_t)ROWS * HDIM * 2);
  char* zbeg  = base + off;                                // zero-init region start
  bf16* hb0a  = (bf16*)carve((size_t)BATCH * HDIM * 2);
  bf16* hb0b  = (bf16*)carve((size_t)BATCH * HDIM * 2);
  bf16* hb1a  = (bf16*)carve((size_t)BATCH * HDIM * 2);
  bf16* hb1b  = (bf16*)carve((size_t)BATCH * HDIM * 2);
  int*  cnt0  = (int*)carve(SEQ * 4);
  int*  cnt1  = (int*)carve(SEQ * 4);
  size_t zlen = (size_t)((base + off) - zbeg);

  const size_t LOGITS = (size_t)ROWS * VOCAB;
  float* logits = out;
  float* h0f = out + LOGITS;
  float* h1f = h0f + (size_t)BATCH * HDIM;
  float* c0f = out + LOGITS + 2 * (size_t)BATCH * HDIM;
  float* c1f = c0f + (size_t)BATCH * HDIM;

  hipMemsetAsync(zbeg, 0, zlen, stream);   // h ping-pong buffers + barrier counters

  const int T = 256;
  embed_kernel<<<(ROWS * E_PAD + T - 1) / T, T, 0, stream>>>(x, emb, eB);
  transpose_cvt_kernel<<<(int)(((long)G4H * E_PAD + T - 1) / T), T, 0, stream>>>(
      w_ih0, wih0T, E_IN, E_PAD, G4H);
  transpose_cvt_kernel<<<(int)(((long)G4H * HDIM + T - 1) / T), T, 0, stream>>>(
      w_hh0, whh0T, HDIM, HDIM, G4H);
  transpose_cvt_kernel<<<(int)(((long)G4H * HDIM + T - 1) / T), T, 0, stream>>>(
      w_ih1, wih1T, HDIM, HDIM, G4H);
  transpose_cvt_kernel<<<(int)(((long)G4H * HDIM + T - 1) / T), T, 0, stream>>>(
      w_hh1, whh1T, HDIM, HDIM, G4H);
  cvt_kernel<<<(int)(((long)VOCAB * HDIM + T - 1) / T), T, 0, stream>>>(
      fc_w, fcwB, (long)VOCAB * HDIM);

  // Layer 0: x_proj = e @ w_ih0 + b0 ; then recurrence
  gemm_bf16_tn_kernel<<<dim3(G4H / 256, ROWS / 128), 256, 0, stream>>>(
      eB, E_PAD, wih0T, E_PAD, b0, xp, G4H, G4H, E_PAD);
  lstm_layer_kernel<<<REC_WGS, 32, 0, stream>>>(
      xp, whh0T, hb0a, hb0b, y0, h0f, c0f, cnt0);

  // Layer 1
  gemm_bf16_tn_kernel<<<dim3(G4H / 256, ROWS / 128), 256, 0, stream>>>(
      y0, HDIM, wih1T, HDIM, b1, xp, G4H, G4H, HDIM);
  lstm_layer_kernel<<<REC_WGS, 32, 0, stream>>>(
      xp, whh1T, hb1a, hb1b, y1, h1f, c1f, cnt1);

  // FC: logits = y1 @ fc_w^T + fc_b   (fc_w is already [N,K] layout)
  gemm_bf16_tn_kernel<<<dim3((VOCAB + 255) / 256, ROWS / 128), 256, 0, stream>>>(
      y1, HDIM, fcwB, HDIM, fc_b, logits, VOCAB, VOCAB, HDIM);
}
